// PlaceCellNetwork_63118839382370
// MI455X (gfx1250) — compile-verified, matrix operand order checked by reading, not run on hardware
//
#include <hip/hip_runtime.h>
#include <hip/hip_bf16.h>

typedef __attribute__((ext_vector_type(16))) _Float16 v16h;
typedef __attribute__((ext_vector_type(8)))  float    v8f;

#define HIDN 20
#define OUTN 10

__device__ __forceinline__ float fast_rcp(float x) { return __builtin_amdgcn_rcpf(x); }
__device__ __forceinline__ float fast_exp(float x) { return __expf(x); } // v_exp_f32 path
__device__ __forceinline__ float fast_tanh(float x) {
    float ax = fabsf(x);
    float e  = fast_exp(-2.0f * ax);
    float r  = (1.0f - e) * fast_rcp(1.0f + e);
    return copysignf(r, x);
}

// Process one 16-row tile. GUARD=true adds row<B checks (tail tile only).
// WMMA calls sit at uniform control flow within the calling wave.
template<bool GUARD>
__device__ __forceinline__ void run_tile(
    const float* __restrict__ x, const float* __restrict__ prev,
    float* __restrict__ out,
    const v16h& a1t0, const v16h& a1t1, const v16h& a2,
    int row, bool hi, int B)
{
    const bool valid = GUARD ? (row < B) : true;

    // ---------------- B1 = [x; 1; prev]^T fragment ----------------
    // B layout (16-bit, 32x16): lane<16 holds column b=l, K {0..7, 16..23};
    // lane>=16 holds column b=l, K {8..15, 24..31}.
    v16h b1 = {};
    if (valid) {
        const float* pr = prev + (size_t)row * OUTN;
        if (!hi) {
            const float2 xv  = *(const float2*)(x + (size_t)row * 2);
            const float2 p01 = *(const float2*)(pr + 0);
            const float2 p23 = *(const float2*)(pr + 2);
            b1[0] = (_Float16)xv.x;  b1[1] = (_Float16)xv.y;  b1[2] = (_Float16)1.0f;
            b1[3] = (_Float16)p01.x; b1[4] = (_Float16)p01.y;
            b1[5] = (_Float16)p23.x; b1[6] = (_Float16)p23.y;
            b1[7] = (_Float16)pr[4];
        } else {
            #pragma unroll
            for (int j = 0; j < 5; ++j) b1[j] = (_Float16)pr[5 + j];
        }
    }

    // ---------------- stage 1: hidden^T = A1 * B1 (f32 accum) ----------------
    v8f c0 = {};
    v8f h0 = __builtin_amdgcn_wmma_f32_16x16x32_f16(false, a1t0, false, b1, (short)0, c0, false, false);
    v8f h1 = __builtin_amdgcn_wmma_f32_16x16x32_f16(false, a1t1, false, b1, (short)0, c0, false, false);

    // tanh + repack: stage-1 D layout (lane<16: M 0..7 / lane>=16: M 8..15, col = l)
    // already matches the B-operand K layout -> no cross-lane transpose needed.
    v16h b2 = {};
    #pragma unroll
    for (int r = 0; r < 8; ++r) b2[r] = (_Float16)fast_tanh(h0[r]);       // h = 0..7 / 8..15
    if (!hi) {
        #pragma unroll
        for (int r = 0; r < 4; ++r) b2[8 + r] = (_Float16)fast_tanh(h1[r]); // h = 16..19
        b2[12] = (_Float16)1.0f;                                            // h = 20 (bias row)
    }

    // ---------------- stage 2: logits^T = A2 * B2 ----------------
    v8f lg = __builtin_amdgcn_wmma_f32_16x16x32_f16(false, a2, false, b2, (short)0, c0, false, false);

    // ---------------- softmax over o, split 8 + 2 across the (l, l+16) lane pair ----------------
    float m;
    if (!hi) {
        m = lg[0];
        #pragma unroll
        for (int r = 1; r < 8; ++r) m = fmaxf(m, lg[r]);
    } else {
        m = fmaxf(lg[0], lg[1]);
    }
    m = fmaxf(m, __shfl_xor(m, 16, 32));

    float e[8];
    #pragma unroll
    for (int r = 0; r < 8; ++r) e[r] = fast_exp(lg[r] - m);
    float s = hi ? (e[0] + e[1])
                 : (e[0] + e[1] + e[2] + e[3] + e[4] + e[5] + e[6] + e[7]);
    s += __shfl_xor(s, 16, 32);
    const float rs = fast_rcp(s);

    if (valid) {
        float* po = out + (size_t)row * OUTN + (hi ? 8 : 0); // 8-byte aligned either way
        if (!hi) {
            float2* p2 = (float2*)po;
            p2[0] = make_float2(e[0] * rs, e[1] * rs);
            p2[1] = make_float2(e[2] * rs, e[3] * rs);
            p2[2] = make_float2(e[4] * rs, e[5] * rs);
            p2[3] = make_float2(e[6] * rs, e[7] * rs);
        } else {
            *(float2*)po = make_float2(e[0] * rs, e[1] * rs);
        }
    }
}

__global__ __launch_bounds__(256) void PlaceCellNetwork_wmma_kernel(
    const float* __restrict__ x,   const float* __restrict__ prev,
    const float* __restrict__ Wh,  const float* __restrict__ bh,
    const float* __restrict__ Wo,  const float* __restrict__ bo,
    const float* __restrict__ Wf,  float* __restrict__ out, int B)
{
    const int tid    = blockIdx.x * blockDim.x + threadIdx.x;
    const int wave   = tid >> 5;
    const int nWaves = (gridDim.x * blockDim.x) >> 5;
    const int lane   = threadIdx.x & 31;
    const int l      = lane & 15;
    const bool hi    = lane >= 16;

    // ---------------- loop-invariant weight fragments (A operands) ----------------
    // Stage-1 A tiles: A1[h,k] = W1[k,h]; k=0,1 -> Wh[:,0..1]; k=2 -> Wh[:,2]+bh (ones row);
    // k=3..12 -> Wf rows 0..9; rest zero. 16-bit A layout: lane<16 holds K {0..7,16..23},
    // lane>=16 holds K {8..15,24..31}, row M = lane&15.
    v16h a1t0 = {}, a1t1 = {}, a2 = {};
    #pragma unroll
    for (int t = 0; t < 2; ++t) {
        const int h = t * 16 + l;
        v16h f = {};
        if (h < HIDN) {
            if (!hi) {
                f[0] = (_Float16)Wh[h * 3 + 0];                  // K=0
                f[1] = (_Float16)Wh[h * 3 + 1];                  // K=1
                f[2] = (_Float16)(Wh[h * 3 + 2] + bh[h]);        // K=2 (ones row)
                #pragma unroll
                for (int j = 0; j < 5; ++j) f[3 + j] = (_Float16)Wf[j * HIDN + h];       // K=3..7
            } else {
                #pragma unroll
                for (int j = 0; j < 5; ++j) f[j] = (_Float16)Wf[(5 + j) * HIDN + h];     // K=8..12
            }
        }
        if (t == 0) a1t0 = f; else a1t1 = f;
    }
    // Stage-2 A: A2[o,h] = Wo[o,h] for h<20, A2[o,20] = bo[o] (bias row), else 0.
    {
        const int o = l;
        if (o < OUTN) {
            if (!hi) {
                #pragma unroll
                for (int j = 0; j < 8; ++j) a2[j]     = (_Float16)Wo[o * HIDN + j];      // K=0..7
                #pragma unroll
                for (int j = 0; j < 4; ++j) a2[8 + j] = (_Float16)Wo[o * HIDN + 16 + j]; // K=16..19
                a2[12] = (_Float16)bo[o];                                                // K=20
            } else {
                #pragma unroll
                for (int j = 0; j < 8; ++j) a2[j]     = (_Float16)Wo[o * HIDN + 8 + j];  // K=8..15
            }
        }
    }

    // ---------------- bulk loop: full 16-row tiles, no bounds checks ----------------
    const int nFull = B >> 4;
    for (int tile = wave; tile < nFull; tile += nWaves) {
        const int row = (tile << 4) + l;

        // Unconditional speculative prefetch (DEV scope) of the next grid-stride tile.
        // OOB addresses are silently dropped by speculative prefetch, so no guard branch.
        const int nrow = row + (nWaves << 4);
        __builtin_prefetch(prev + (size_t)nrow * OUTN, 0, 1);
        __builtin_prefetch(x    + (size_t)nrow * 2,    0, 1);

        run_tile<false>(x, prev, out, a1t0, a1t1, a2, row, hi, B);
    }

    // ---------------- tail tile (B % 16 rows), handled by wave 0 only ----------------
    if ((B & 15) && wave == 0) {
        const int row = (nFull << 4) + l;
        run_tile<true>(x, prev, out, a1t0, a1t1, a2, row, hi, B);
    }
}

extern "C" void kernel_launch(void* const* d_in, const int* in_sizes, int n_in,
                              void* d_out, int out_size, void* d_ws, size_t ws_size,
                              hipStream_t stream) {
    const float* x    = (const float*)d_in[0];
    const float* prev = (const float*)d_in[1];
    const float* Wh   = (const float*)d_in[2];
    const float* bh   = (const float*)d_in[3];
    const float* Wo   = (const float*)d_in[4];
    const float* bo   = (const float*)d_in[5];
    const float* Wf   = (const float*)d_in[6];
    float* out = (float*)d_out;

    const int B = in_sizes[0] / 2;   // x is [B, 2]

    const int threads = 256;         // 8 wave32 per block
    const int blocks  = 1024;        // 8192 waves, grid-stride over B/16 = 250k tiles
    PlaceCellNetwork_wmma_kernel<<<blocks, threads, 0, stream>>>(
        x, prev, Wh, bh, Wo, bo, Wf, out, B);
}
